// GATLinkPredictor_5016521801944
// MI455X (gfx1250) — matrix operand order repaired and checked by reading, not compiled, not run on hardware
//
#include <hip/hip_runtime.h>
#include <hip/hip_bf16.h>
#include <limits.h>

typedef __attribute__((ext_vector_type(16))) __bf16 v16bf;
typedef __attribute__((ext_vector_type(8)))  __bf16 v8bf;
typedef __attribute__((ext_vector_type(8)))  float  v8f;

// ---------- helpers ----------
__device__ inline unsigned short f2bf(float f) {
    unsigned int u = __float_as_uint(f);
    unsigned int r = (u + 0x7fffu + ((u >> 16) & 1u)) >> 16;   // RNE
    return (unsigned short)r;
}
__device__ inline int f2ord(float f) {            // monotonic float->int key
    int b = __float_as_int(f);
    return b >= 0 ? b : (b ^ 0x7fffffff);
}
__device__ inline float ord2f(int k) {
    return __int_as_float(k >= 0 ? k : (k ^ 0x7fffffff));
}
// A fragment: 16x32 bf16 tile, row = lane%16. a[0..7]=K(8*half..), a[8..15]=K(16+8*half..)
__device__ inline v16bf load_a_frag(const unsigned short* rowbase, int half) {
    v8bf lo = *(const v8bf*)(rowbase + 8 * half);
    v8bf hi = *(const v8bf*)(rowbase + 16 + 8 * half);
    return __builtin_shufflevector(lo, hi, 0,1,2,3,4,5,6,7,8,9,10,11,12,13,14,15);
}
// B fragment: 32x16 bf16 (B pre-transposed, column-major): b[0..7]=K(8*half..), b[8..15]=K(16+8*half..)
__device__ inline v16bf load_b_frag(const unsigned short* colbase, int half) {
    v8bf lo = *(const v8bf*)(colbase + 8 * half);
    v8bf hi = *(const v8bf*)(colbase + 16 + 8 * half);
    return __builtin_shufflevector(lo, hi, 0,1,2,3,4,5,6,7,8,9,10,11,12,13,14,15);
}

// ---------- elementwise / prep kernels ----------
__global__ void k_f32_to_bf16(const float* __restrict__ s, unsigned short* __restrict__ d, long long n) {
    long long t = (long long)blockIdx.x * blockDim.x + threadIdx.x;
    if (t < n) d[t] = f2bf(s[t]);
}
// src: [K][M] row-major f32 -> dst: [M][K] bf16
__global__ void k_transpose_bf16(const float* __restrict__ src, unsigned short* __restrict__ dst,
                                 int K, int M) {
    int t = blockIdx.x * blockDim.x + threadIdx.x;
    if (t >= K * M) return;
    int k = t / M, m = t % M;
    dst[(long long)m * K + k] = f2bf(src[t]);
}
__global__ void k_fill_i32(int* __restrict__ p, int v, long long n) {
    long long t = (long long)blockIdx.x * blockDim.x + threadIdx.x;
    if (t < n) p[t] = v;
}
// y = act(in + bias[col]) -> bf16.  act: 1 = ELU, 0 = identity
__global__ void k_bias_act_bf16(const float* __restrict__ in, const float* __restrict__ bias,
                                unsigned short* __restrict__ ob, long long rows, int cols, int do_elu) {
    long long t = (long long)blockIdx.x * blockDim.x + threadIdx.x;
    if (t >= rows * cols) return;
    int c = (int)(t % cols);
    float v = in[t] + bias[c];
    if (do_elu) v = v > 0.0f ? v : (__expf(v) - 1.0f);
    ob[t] = f2bf(v);
}

// ---------- WMMA GEMM with A-fragment reuse ----------
// C[rows,cols] = A[rows,128](bf16) x Bt[cols,128](bf16).
// One wave owns one 16-row strip; the full K=128 A strip lives in registers
// (4 x v16bf = 32 VGPRs) and is reused across every 16-wide column tile.
__global__ void k_gemm_bf16_k128(const unsigned short* __restrict__ A,
                                 const unsigned short* __restrict__ Bt,
                                 float* __restrict__ C, int rows, int cols) {
    const int KSTEPS = 4;                          // K = 128
    const int lane = threadIdx.x & 31;
    const int wid  = (int)((blockIdx.x * (long long)blockDim.x + threadIdx.x) >> 5);
    const int rowTiles = (rows + 15) >> 4;
    if (wid >= rowTiles) return;                   // wave-uniform exit
    const int half = lane >> 4;
    const int arow = min(wid * 16 + (lane & 15), rows - 1);
    const unsigned short* ap = A + (long long)arow * 128;
    v16bf afrag[KSTEPS];
    #pragma unroll
    for (int k = 0; k < KSTEPS; k++) afrag[k] = load_a_frag(ap + 32 * k, half);
    const int colTiles = cols >> 4;
    const bool fullTile = (wid * 16 + 16 <= rows); // wave-uniform
    for (int tc = 0; tc < colTiles; tc++) {
        const unsigned short* bp = Bt + (long long)(tc * 16 + (lane & 15)) * 128;
        v8f c = {};
        #pragma unroll
        for (int k = 0; k < KSTEPS; k++) {
            v16bf b = load_b_frag(bp + 32 * k, half);
            c = __builtin_amdgcn_wmma_f32_16x16x32_bf16(false, afrag[k], false, b,
                                                        (short)0, c, false, false);
        }
        float* cp = C + (long long)(wid * 16 + 8 * half) * cols + tc * 16 + (lane & 15);
        if (fullTile) {                            // no EXEC manipulation on hot path
            #pragma unroll
            for (int i = 0; i < 8; i++) cp[(long long)i * cols] = c[i];
        } else {
            #pragma unroll
            for (int i = 0; i < 8; i++)
                if (wid * 16 + 8 * half + i < rows) cp[(long long)i * cols] = c[i];
        }
    }
}

// ---------- attention kernels ----------
// per-(node, head) dot products with att vectors
__global__ void k_node_alpha(const float* __restrict__ h, const float* __restrict__ a_s,
                             const float* __restrict__ a_d, float* __restrict__ as,
                             float* __restrict__ ad, int N, int H, int C) {
    int t = blockIdx.x * blockDim.x + threadIdx.x;
    if (t >= N * H) return;
    int n = t / H, hh = t % H;
    const float* hp  = h + (long long)n * H * C + hh * C;
    const float* asp = a_s + hh * C;
    const float* adp = a_d + hh * C;
    float s1 = 0.f, s2 = 0.f;
    for (int c = 0; c < C; c++) { float v = hp[c]; s1 += v * asp[c]; s2 += v * adp[c]; }
    as[t] = s1; ad[t] = s2;
}
__device__ inline void edge_sd(const int* ei, int E, unsigned int e, int& s, int& d) {
    if ((int)e < E) { s = ei[e]; d = ei[E + e]; } else { s = d = (int)e - E; }
}
__global__ void k_edge_max(const int* __restrict__ ei, int E, int N,
                           const float* __restrict__ as, const float* __restrict__ ad,
                           int* __restrict__ m, int H) {
    unsigned int t = blockIdx.x * blockDim.x + threadIdx.x;
    unsigned int tot = (unsigned int)(E + N) * H;
    if (t >= tot) return;
    unsigned int e = t / H; int hh = t % H;
    int s, d; edge_sd(ei, E, e, s, d);
    float v = as[s * H + hh] + ad[d * H + hh];
    v = v > 0.f ? v : 0.2f * v;
    atomicMax(&m[d * H + hh], f2ord(v));
}
__global__ void k_edge_sum(const int* __restrict__ ei, int E, int N,
                           const float* __restrict__ as, const float* __restrict__ ad,
                           const int* __restrict__ m, float* __restrict__ sb, int H) {
    unsigned int t = blockIdx.x * blockDim.x + threadIdx.x;
    unsigned int tot = (unsigned int)(E + N) * H;
    if (t >= tot) return;
    unsigned int e = t / H; int hh = t % H;
    int s, d; edge_sd(ei, E, e, s, d);
    float v = as[s * H + hh] + ad[d * H + hh];
    v = v > 0.f ? v : 0.2f * v;
    atomicAdd(&sb[d * H + hh], __expf(v - ord2f(m[d * H + hh])));
}
__global__ void k_edge_aggr(const int* __restrict__ ei, int E, int N,
                            const float* __restrict__ as, const float* __restrict__ ad,
                            const int* __restrict__ m, const float* __restrict__ sb,
                            const float* __restrict__ h, float* __restrict__ out, int H, int C) {
    unsigned int t = blockIdx.x * blockDim.x + threadIdx.x;
    unsigned int HC = (unsigned int)H * C;
    unsigned int tot = (unsigned int)(E + N) * HC;
    if (t >= tot) return;
    unsigned int e = t / HC; unsigned int r = t % HC; int hh = r / C;
    int s, d; edge_sd(ei, E, e, s, d);
    float v = as[s * H + hh] + ad[d * H + hh];
    v = v > 0.f ? v : 0.2f * v;
    float a = __expf(v - ord2f(m[d * H + hh])) / (sb[d * H + hh] + 1e-16f);
    atomicAdd(&out[(long long)d * HC + r], h[(long long)s * HC + r] * a);
}

// ---------- fused pair MLP: relu(concat(h[ps],h[pd]) @ Wm1 + bm1) @ Wm2 + bm2 ----------
__global__ void k_pair_mlp(const unsigned short* __restrict__ hb /*[N,64] bf16*/,
                           const int* __restrict__ pairs /*ps[0..P), pd[P..2P)*/,
                           const unsigned short* __restrict__ Wt /*[32][128] bf16*/,
                           const float* __restrict__ bm1, const float* __restrict__ Wm2,
                           const float* __restrict__ bm2, float* __restrict__ out, int P) {
    const int lane = threadIdx.x & 31;
    const int wid  = (int)((blockIdx.x * (long long)blockDim.x + threadIdx.x) >> 5);
    const int tiles = (P + 15) >> 4;
    if (wid >= tiles) return;                      // wave-uniform exit
    const int half = lane >> 4;
    const int r  = min(wid * 16 + (lane & 15), P - 1);
    const unsigned short* hs = hb + (long long)pairs[r] * 64;
    const unsigned short* hd = hb + (long long)pairs[P + r] * 64;
    v8f c0 = {}, c1 = {};
    #pragma unroll
    for (int k0 = 0; k0 < 128; k0 += 32) {
        const unsigned short* src = (k0 < 64) ? (hs + k0) : (hd + (k0 - 64));
        v16bf a  = load_a_frag(src, half);
        v16bf b0 = load_b_frag(Wt + (long long)(lane & 15) * 128 + k0, half);
        v16bf b1 = load_b_frag(Wt + (long long)(16 + (lane & 15)) * 128 + k0, half);
        c0 = __builtin_amdgcn_wmma_f32_16x16x32_bf16(false, a, false, b0, (short)0, c0, false, false);
        c1 = __builtin_amdgcn_wmma_f32_16x16x32_bf16(false, a, false, b1, (short)0, c1, false, false);
    }
    const int n = lane & 15;
    const float w0 = Wm2[n],      w1 = Wm2[16 + n];
    const float g0 = bm1[n],      g1 = bm1[16 + n];
    float res[8];
    #pragma unroll
    for (int i = 0; i < 8; i++) {
        float h0 = c0[i] + g0; h0 = h0 > 0.f ? h0 : 0.f;
        float h1 = c1[i] + g1; h1 = h1 > 0.f ? h1 : 0.f;
        res[i] = h0 * w0 + h1 * w1;
    }
    #pragma unroll
    for (int msk = 1; msk < 16; msk <<= 1)
        #pragma unroll
        for (int i = 0; i < 8; i++) res[i] += __shfl_xor(res[i], msk, 32);
    const bool fullTile = (wid * 16 + 16 <= P);    // wave-uniform
    if ((lane & 15) == 0) {
        float bb = bm2[0];
        float* op = out + wid * 16 + 8 * half;
        if (fullTile) {
            #pragma unroll
            for (int i = 0; i < 8; i++) op[i] = res[i] + bb;
        } else {
            #pragma unroll
            for (int i = 0; i < 8; i++)
                if (wid * 16 + 8 * half + i < P) op[i] = res[i] + bb;
        }
    }
}

// ---------- host ----------
static inline unsigned int nblk(long long n, int b) { return (unsigned int)((n + b - 1) / b); }

extern "C" void kernel_launch(void* const* d_in, const int* in_sizes, int n_in,
                              void* d_out, int out_size, void* d_ws, size_t ws_size,
                              hipStream_t stream) {
    const float* x      = (const float*)d_in[0];
    const int*   ei     = (const int*)d_in[1];
    const int*   pairs  = (const int*)d_in[2];
    const float* W1     = (const float*)d_in[3];
    const float* att_s1 = (const float*)d_in[4];
    const float* att_d1 = (const float*)d_in[5];
    const float* b1     = (const float*)d_in[6];
    const float* W2     = (const float*)d_in[7];
    const float* att_s2 = (const float*)d_in[8];
    const float* att_d2 = (const float*)d_in[9];
    const float* b2     = (const float*)d_in[10];
    const float* Wm1    = (const float*)d_in[11];
    const float* bm1    = (const float*)d_in[12];
    const float* Wm2    = (const float*)d_in[13];
    const float* bm2    = (const float*)d_in[14];
    float* out = (float*)d_out;

    const int N = in_sizes[0] / 128;
    const int E = in_sizes[1] / 2;
    const int P = in_sizes[2] / 2;
    const int Ep = E + N;                         // with self-loops

    // workspace layout (256B aligned), with reuse
    size_t off = 0;
    auto alloc = [&](size_t bytes) { size_t r = off; off += (bytes + 255) & ~(size_t)255; return r; };
    char* ws = (char*)d_ws;
    unsigned short* xb   = (unsigned short*)(ws + alloc((size_t)N * 128 * 2)); // later: h2b
    float*          h1   = (float*)(ws + alloc((size_t)N * 128 * 4));          // later: h2
    float*          out1 = (float*)(ws + alloc((size_t)N * 128 * 4));          // later: out2
    unsigned short* h1b  = (unsigned short*)(ws + alloc((size_t)N * 128 * 2));
    float*          asb  = (float*)(ws + alloc((size_t)N * 4 * 4));
    float*          adb  = (float*)(ws + alloc((size_t)N * 4 * 4));
    int*            mb   = (int*)  (ws + alloc((size_t)N * 4 * 4));
    float*          sb   = (float*)(ws + alloc((size_t)N * 4 * 4));
    unsigned short* W1t  = (unsigned short*)(ws + alloc(128 * 128 * 2));
    unsigned short* W2t  = (unsigned short*)(ws + alloc(64 * 128 * 2));
    unsigned short* Wm1t = (unsigned short*)(ws + alloc(32 * 128 * 2));
    float*          h2   = h1;     // [N,64]
    float*          out2 = out1;   // [N,64]
    unsigned short* h2b  = xb;     // [N,64]

    const int B = 256;
    const int rowT = (N + 15) / 16;

    // prep: bf16 conversions and weight transposes
    k_f32_to_bf16<<<nblk((long long)N * 128, B), B, 0, stream>>>(x, xb, (long long)N * 128);
    k_transpose_bf16<<<nblk(128 * 128, B), B, 0, stream>>>(W1, W1t, 128, 128);
    k_transpose_bf16<<<nblk(128 * 64, B), B, 0, stream>>>(W2, W2t, 128, 64);
    k_transpose_bf16<<<nblk(128 * 32, B), B, 0, stream>>>(Wm1, Wm1t, 128, 32);

    // ---- GAT layer 1 (H=4, C=32) ----
    k_gemm_bf16_k128<<<nblk((long long)rowT * 32, B), B, 0, stream>>>(xb, W1t, h1, N, 128);
    k_node_alpha<<<nblk((long long)N * 4, B), B, 0, stream>>>(h1, att_s1, att_d1, asb, adb, N, 4, 32);
    k_fill_i32<<<nblk((long long)N * 4, B), B, 0, stream>>>(mb, INT_MIN, (long long)N * 4);
    hipMemsetAsync(sb, 0, (size_t)N * 4 * 4, stream);
    hipMemsetAsync(out1, 0, (size_t)N * 128 * 4, stream);
    k_edge_max <<<nblk((long long)Ep * 4, B), B, 0, stream>>>(ei, E, N, asb, adb, mb, 4);
    k_edge_sum <<<nblk((long long)Ep * 4, B), B, 0, stream>>>(ei, E, N, asb, adb, mb, sb, 4);
    k_edge_aggr<<<nblk((long long)Ep * 128, B), B, 0, stream>>>(ei, E, N, asb, adb, mb, sb, h1, out1, 4, 32);
    k_bias_act_bf16<<<nblk((long long)N * 128, B), B, 0, stream>>>(out1, b1, h1b, N, 128, 1);

    // ---- GAT layer 2 (H=1, C=64) ----
    k_gemm_bf16_k128<<<nblk((long long)rowT * 32, B), B, 0, stream>>>(h1b, W2t, h2, N, 64);
    k_node_alpha<<<nblk((long long)N, B), B, 0, stream>>>(h2, att_s2, att_d2, asb, adb, N, 1, 64);
    k_fill_i32<<<nblk((long long)N, B), B, 0, stream>>>(mb, INT_MIN, (long long)N);
    hipMemsetAsync(sb, 0, (size_t)N * 4, stream);
    hipMemsetAsync(out2, 0, (size_t)N * 64 * 4, stream);
    k_edge_max <<<nblk((long long)Ep, B), B, 0, stream>>>(ei, E, N, asb, adb, mb, 1);
    k_edge_sum <<<nblk((long long)Ep, B), B, 0, stream>>>(ei, E, N, asb, adb, mb, sb, 1);
    k_edge_aggr<<<nblk((long long)Ep * 64, B), B, 0, stream>>>(ei, E, N, asb, adb, mb, sb, h2, out2, 1, 64);
    k_bias_act_bf16<<<nblk((long long)N * 64, B), B, 0, stream>>>(out2, b2, h2b, N, 64, 0);

    // ---- fused pair MLP (WMMA) ----
    {
        int tiles = (P + 15) / 16;
        k_pair_mlp<<<nblk((long long)tiles * 32, B), B, 0, stream>>>(h2b, pairs, Wm1t, bm1, Wm2, bm2, out, P);
    }
}
